// uniLSTM_72550587564392
// MI455X (gfx1250) — matrix operand order repaired and checked by reading, not compiled, not run on hardware
//
#include <hip/hip_runtime.h>

// ---------------------------------------------------------------------------
// Unidirectional LSTM on MI455X (gfx1250, wave32, WMMA).
//
// Strategy:
//   * Prep kernels (fully parallel):
//       - y = bp broadcast (output accumulator init; d_out is poisoned)
//       - pack Wf/Wi/Wo/Wc a_prev-rows (512x128 per WG) into bf16 WMMA
//         B-fragments, LDS-resident during the sequential phase
//       - pack the x-rows (300->320 padded) into bf16 B-fragments (L2-hot)
//       - embedding lookup packed directly into bf16 WMMA A-fragments per step
//       - pack Wp into B-fragments for the per-step projection
//   * Sequential kernel: 16 workgroups (one WGP each), each owns 32 hidden
//     columns (= 128 gate columns). Per step:
//       TDM a_prev frags (64KB) + x frags (40KB) -> LDS, s_wait_tensorcnt
//       26 K-tiles of v_wmma_f32_16x16x32_bf16 (8 waves x 4 M-tiles)
//       gate nonlinearities, c-state in LDS, a_next -> bf16 A-frags (global)
//       projection partial (a_slice @ Wp_slice) via WMMA + atomicAdd into y
//       cluster barrier (+ deterministic atomic arrive counter) across WGs
// ---------------------------------------------------------------------------

#define B_    64
#define S_    256
#define DIN_  300
#define HID_  512
#define OUT_  32
#define NWG   16
#define KT_A  16          // 512/32 K-tiles for the a_prev part
#define KT_X  10          // 320/32 K-tiles for the x part (padded from 300)

typedef unsigned int u32;
typedef unsigned long long u64;
typedef float  v8f   __attribute__((ext_vector_type(8)));
typedef __bf16 v16bf __attribute__((ext_vector_type(16)));
typedef u32    v4u   __attribute__((ext_vector_type(4)));
typedef int    v4i   __attribute__((ext_vector_type(4)));
typedef int    v8i   __attribute__((ext_vector_type(8)));

// ---- workspace layout (bytes) ----
#define OFF_BA    ((size_t)0)                       // [16 wg][16 kt][8 nt][1024]
#define SZ_BA     ((size_t)16*16*8*1024)            // 2 MiB
#define OFF_BX    (OFF_BA + SZ_BA)                  // [16 wg][10 kt][8 nt][1024]
#define SZ_BX     ((size_t)16*10*8*1024)            // 1.25 MiB
#define OFF_WP    (OFF_BX + SZ_BX)                  // [16 wg][2 nt][1024]
#define SZ_WP     ((size_t)16*2*1024)
#define OFF_AX    (OFF_WP + SZ_WP)                  // [256 s][10 kt][4 mt][1024]
#define SZ_AX     ((size_t)256*10*4*1024)           // 10 MiB
#define OFF_ABUF  (OFF_AX + SZ_AX)                  // double buffer [2][16kt][4mt][1024]
#define SZ_ABUF   ((size_t)2*16*4*1024)
#define OFF_SYNC  (OFF_ABUF + SZ_ABUF)              // 256 arrive counters

// ---- dynamic LDS layout (bytes) ----
#define L_BA    0                                   // 131072: resident Wa frags
#define L_AST   131072                              // 106496: A frags [26kt][4mt][1024]
#define L_GATES (L_AST)                             // alias after GEMM: [4][64][32] f32
#define L_ASTG  (L_AST + 32768)                     // alias: a_next staging [64][32] f32
#define L_AFRG  (L_AST + 40960)                     // alias: a_next bf16 frags (4 KB)
#define L_C     (L_AST + 106496)                    // 8192: c state [64][32] f32
#define L_WP    (L_C + 8192)                        // 2048: Wp frags
#define LDS_TOTAL (L_WP + 2048)                     // 247808 bytes

// ---------------------------------------------------------------------------
__device__ __forceinline__ u32 packbf2(float lo, float hi) {
  u32 a = __float_as_uint(lo), b = __float_as_uint(hi);
  a = (a + 0x7FFFu + ((a >> 16) & 1u)) >> 16;       // round-to-nearest-even
  b = (b + 0x7FFFu + ((b >> 16) & 1u)) >> 16;
  return (a & 0xFFFFu) | (b << 16);
}

__device__ __forceinline__ float sig_f(float x) { return 1.f / (1.f + __expf(-x)); }
__device__ __forceinline__ float tanh_f(float x) { return 1.f - 2.f / (1.f + __expf(2.f * x)); }

__device__ __forceinline__ v8f wmma_bf16(v16bf a, v16bf b, v8f c) {
  return __builtin_amdgcn_wmma_f32_16x16x32_bf16(false, a, false, b, (short)0, c,
                                                 false, false);
}

// One-row (contiguous) TDM descriptor: ndw dwords global -> LDS.  (D# per ISA §8)
__device__ __forceinline__ void tdm_load_1d(void* ldsPtr, const void* gptr, u32 ndw) {
  u32 ldsOff = (u32)(u64)(uintptr_t)ldsPtr;         // addr[31:0] == LDS byte offset
  u64 ga = (u64)(uintptr_t)gptr;
  v4u g0;
  g0.x = 1u;                                        // count=1 (valid descriptor)
  g0.y = ldsOff;                                    // lds_addr
  g0.z = (u32)ga;                                   // global_addr lo
  g0.w = ((u32)(ga >> 32) & 0x01FFFFFFu) | (2u << 30);  // global_addr hi | type=2
  v8i g1;
  g1[0] = (int)(2u << 16);                          // data_size = 4 bytes
  g1[1] = (int)((ndw & 0xFFFFu) << 16);             // tensor_dim0 lo16
  g1[2] = (int)((ndw >> 16) | (1u << 16));          // tensor_dim0 hi16 | tensor_dim1=1
  g1[3] = (int)((ndw & 0xFFFFu) << 16);             // tile_dim0 = ndw (<= 65535)
  g1[4] = 1;                                        // tile_dim1 = 1
  g1[5] = (int)ndw;                                 // tensor_dim0_stride lo32
  g1[6] = (int)((ndw & 0xFFFFu) << 16);             // stride hi16=0 | dim1_stride lo16
  g1[7] = (int)(ndw >> 16);                         // dim1_stride hi
  v4i gz = {0, 0, 0, 0};
#if defined(__clang_major__) && (__clang_major__ >= 23)
  v8i gz8 = {0, 0, 0, 0, 0, 0, 0, 0};
  __builtin_amdgcn_tensor_load_to_lds(g0, g1, gz, gz, gz8, 0);
#else
  __builtin_amdgcn_tensor_load_to_lds(g0, g1, gz, gz, 0);
#endif
}

// ---------------------------------------------------------------------------
// Prep kernels
// ---------------------------------------------------------------------------
__global__ void lstm_init_y(float* __restrict__ y, const float* __restrict__ bp) {
  int i = blockIdx.x * blockDim.x + threadIdx.x;
  if (i < B_ * S_ * OUT_) y[i] = bp[i & (OUT_ - 1)];
}

__global__ void lstm_zero_ws(char* __restrict__ ws) {
  int i = blockIdx.x * blockDim.x + threadIdx.x;
  if (i < 256) ((u32*)(ws + OFF_SYNC))[i] = 0u;                 // arrive counters
  else if (i < 256 + 16384) ((u32*)(ws + OFF_ABUF))[i - 256] = 0u;  // a_prev buf[0]=0
}

// Pack Wa (rows 0..511 of each gate weight) into bf16 B-fragments.
// B frag (32x16, 16-bit), contiguous K as packed here: lanes 0-15 K=0..15, lanes 16-31 K=16..31; dword v
// packs rows (2v, 2v+1) of that half for column n = lane&15.
__global__ void lstm_pack_ba(char* __restrict__ ws,
                             const float* __restrict__ Wf, const float* __restrict__ Wi,
                             const float* __restrict__ Wo, const float* __restrict__ Wc) {
  int q = blockIdx.x * blockDim.x + threadIdx.x;
  if (q >= (int)(SZ_BA / 4)) return;
  int f = q >> 8, p = q & 255, lane = p >> 3, v = p & 7;
  int nt = f & 7, kt = (f >> 3) & 15, wg = f >> 7;
  int n = lane & 15, khalf = (lane >= 16) ? 16 : 0;
  int nloc = nt * 16 + n, gate = nloc >> 5, j = wg * 32 + (nloc & 31);
  const float* W = (gate == 0) ? Wf : (gate == 1) ? Wi : (gate == 2) ? Wo : Wc;
  int k = kt * 32 + khalf + 2 * v;
  ((u32*)(ws + OFF_BA))[q] = packbf2(W[(size_t)k * HID_ + j], W[(size_t)(k + 1) * HID_ + j]);
}

// Pack Wx (rows 512..811, zero-padded to 832).
__global__ void lstm_pack_bx(char* __restrict__ ws,
                             const float* __restrict__ Wf, const float* __restrict__ Wi,
                             const float* __restrict__ Wo, const float* __restrict__ Wc) {
  int q = blockIdx.x * blockDim.x + threadIdx.x;
  if (q >= (int)(SZ_BX / 4)) return;
  int f = q >> 8, p = q & 255, lane = p >> 3, v = p & 7;
  int nt = f & 7, kt = (f >> 3) % 10, wg = f / 80;
  int n = lane & 15, khalf = (lane >= 16) ? 16 : 0;
  int nloc = nt * 16 + n, gate = nloc >> 5, j = wg * 32 + (nloc & 31);
  const float* W = (gate == 0) ? Wf : (gate == 1) ? Wi : (gate == 2) ? Wo : Wc;
  int k = HID_ + kt * 32 + khalf + 2 * v;
  float lo = (k < HID_ + DIN_) ? W[(size_t)k * HID_ + j] : 0.f;
  float hi = (k + 1 < HID_ + DIN_) ? W[(size_t)(k + 1) * HID_ + j] : 0.f;
  ((u32*)(ws + OFF_BX))[q] = packbf2(lo, hi);
}

// Pack Wp (HID x OUT): per WG a 32x32 B tile -> 2 N-tiles.
__global__ void lstm_pack_wp(char* __restrict__ ws, const float* __restrict__ Wp) {
  int q = blockIdx.x * blockDim.x + threadIdx.x;
  if (q >= (int)(SZ_WP / 4)) return;
  int f = q >> 8, p = q & 255, lane = p >> 3, v = p & 7;
  int nt = f & 1, wg = f >> 1;
  int o = nt * 16 + (lane & 15), khalf = (lane >= 16) ? 16 : 0;
  int row = wg * 32 + khalf + 2 * v;
  ((u32*)(ws + OFF_WP))[q] = packbf2(Wp[(size_t)row * OUT_ + o],
                                     Wp[(size_t)(row + 1) * OUT_ + o]);
}

// Embedding lookup packed straight into bf16 A-fragments per step.
// A frag (16x32, 16-bit), contiguous K as packed here: lane l -> row M=l&15; lanes<16 hold K 0..15 in
// v0..7; lanes>=16 hold K 16..31.
__global__ void lstm_pack_ax(char* __restrict__ ws, const int* __restrict__ x,
                             const float* __restrict__ emb) {
  int q = blockIdx.x * blockDim.x + threadIdx.x;
  if (q >= (int)(SZ_AX / 4)) return;
  int f = q >> 8, p = q & 255, lane = p >> 3, v = p & 7;
  int mt = f & 3, kt = (f >> 2) % 10, s = f / 40;
  int b = mt * 16 + (lane & 15);
  int base = (lane >= 16) ? 16 : 0;
  int K = base + 2 * v;
  int d = kt * 32 + K;
  const float* row = emb + (size_t)x[b * S_ + s] * DIN_;
  float lo = (d < DIN_) ? row[d] : 0.f;
  float hi = (d + 1 < DIN_) ? row[d + 1] : 0.f;
  ((u32*)(ws + OFF_AX))[q] = packbf2(lo, hi);
}

// ---------------------------------------------------------------------------
// Sequential recurrence: 16 cooperating workgroups, 256 threads (8 waves) each
// ---------------------------------------------------------------------------
__global__ __launch_bounds__(256) void lstm_seq(
    const float* __restrict__ bfv, const float* __restrict__ biv,
    const float* __restrict__ bov, const float* __restrict__ bcv,
    float* __restrict__ y, char* __restrict__ ws) {
  extern __shared__ char sm[];
  const int g = blockIdx.x;
  const int t = threadIdx.x;
  const int lane = t & 31;
  const int w = t >> 5;

  // ---- prologue: weights -> LDS via TDM; zero c state; biases -> regs ----
  if (t < 32) {
    tdm_load_1d(sm + L_BA, ws + OFF_BA + (size_t)g * (16 * 8 * 1024), (16 * 8 * 1024) / 4);
    tdm_load_1d(sm + L_WP, ws + OFF_WP + (size_t)g * 2048, 2048 / 4);
    __builtin_amdgcn_s_wait_tensorcnt(0);
  }
  float* cSt = (float*)(sm + L_C);
  for (int i = t; i < 64 * 32; i += 256) cSt[i] = 0.f;
  const int col = t & 31;
  const int j = g * 32 + col;
  const float rbf = bfv[j], rbi = biv[j], rbo = bov[j], rbc = bcv[j];
  u32* arrive = (u32*)(ws + OFF_SYNC);
  const char* bxG = ws + OFF_BX + (size_t)g * (10 * 8 * 1024);
  __syncthreads();

  for (int s = 0; s < S_; ++s) {
    // ---- stage A fragments (a_prev 64 KB + x part 40 KB) via TDM ----
    if (t < 32) {
      tdm_load_1d(sm + L_AST, ws + OFF_ABUF + (size_t)(s & 1) * 65536, 65536 / 4);
      tdm_load_1d(sm + L_AST + 65536, ws + OFF_AX + (size_t)s * (10 * 4 * 1024), 40960 / 4);
      __builtin_amdgcn_s_wait_tensorcnt(0);
    }
    __syncthreads();

    // ---- gate GEMM: (64 x 832) @ (832 x 128), wave w owns 16 gate columns ----
    v8f acc[4] = {{0.f}, {0.f}, {0.f}, {0.f}};
    const char* baL = sm + L_BA;
    const char* astL = sm + L_AST;
#pragma unroll 2
    for (int kt = 0; kt < KT_A; ++kt) {
      v16bf bfr = *(const v16bf*)(baL + (size_t)(kt * 8 + w) * 1024 + lane * 32);
      v16bf a0 = *(const v16bf*)(astL + (size_t)(kt * 4 + 0) * 1024 + lane * 32);
      v16bf a1 = *(const v16bf*)(astL + (size_t)(kt * 4 + 1) * 1024 + lane * 32);
      v16bf a2 = *(const v16bf*)(astL + (size_t)(kt * 4 + 2) * 1024 + lane * 32);
      v16bf a3 = *(const v16bf*)(astL + (size_t)(kt * 4 + 3) * 1024 + lane * 32);
      acc[0] = wmma_bf16(a0, bfr, acc[0]);
      acc[1] = wmma_bf16(a1, bfr, acc[1]);
      acc[2] = wmma_bf16(a2, bfr, acc[2]);
      acc[3] = wmma_bf16(a3, bfr, acc[3]);
    }
#pragma unroll 2
    for (int kt = 0; kt < KT_X; ++kt) {
      v16bf bfr = *(const v16bf*)(bxG + (size_t)(kt * 8 + w) * 1024 + lane * 32);
      const char* ab = astL + 65536 + (size_t)kt * 4096;
      v16bf a0 = *(const v16bf*)(ab + 0 * 1024 + lane * 32);
      v16bf a1 = *(const v16bf*)(ab + 1 * 1024 + lane * 32);
      v16bf a2 = *(const v16bf*)(ab + 2 * 1024 + lane * 32);
      v16bf a3 = *(const v16bf*)(ab + 3 * 1024 + lane * 32);
      acc[0] = wmma_bf16(a0, bfr, acc[0]);
      acc[1] = wmma_bf16(a1, bfr, acc[1]);
      acc[2] = wmma_bf16(a2, bfr, acc[2]);
      acc[3] = wmma_bf16(a3, bfr, acc[3]);
    }
    __syncthreads();  // everyone done reading A staging; OK to alias region

    // ---- scatter D tiles to LDS gates[4][64][32] (f32 C/D layout, §7.12.2) ----
    {
      float* gates = (float*)(sm + L_GATES);
      int gate = w >> 1;
      int gcol = ((w & 1) << 4) | (lane & 15);
      int mb = (lane >> 4) << 3;  // 0 or 8
#pragma unroll
      for (int mt = 0; mt < 4; ++mt)
#pragma unroll
        for (int r = 0; r < 8; ++r)
          gates[gate * 2048 + (mt * 16 + mb + r) * 32 + gcol] = acc[mt][r];
    }
    __syncthreads();

    // ---- elementwise LSTM cell update (thread owns col, 8 batch rows) ----
    {
      float* gates = (float*)(sm + L_GATES);
      float* aStg = (float*)(sm + L_ASTG);
      int brow = t >> 5;
#pragma unroll
      for (int r = 0; r < 8; ++r) {
        int idx = (brow * 8 + r) * 32 + col;
        float ft = sig_f(gates[0 * 2048 + idx] + rbf);
        float it = sig_f(gates[1 * 2048 + idx] + rbi);
        float ot = sig_f(gates[2 * 2048 + idx] + rbo);
        float cc = tanh_f(gates[3 * 2048 + idx] + rbc);
        float cn = ft * cSt[idx] + it * cc;
        cSt[idx] = cn;
        aStg[idx] = ot * tanh_f(cn);
      }
    }
    __syncthreads();

    // ---- assemble a_next bf16 A-fragments: LDS copy + global (next step) ----
    {
      const float* aStg = (const float*)(sm + L_ASTG);
      int mt = t >> 6, ln = (t >> 1) & 31, half = t & 1;
      int b = mt * 16 + (ln & 15);
      int kb = (ln >= 16) ? 16 : 0;
      uint4 dv;
      u32 dws[4];
#pragma unroll
      for (int i = 0; i < 4; ++i) {
        int K = kb + 8 * half + 2 * i;
        dws[i] = packbf2(aStg[b * 32 + K], aStg[b * 32 + K + 1]);
      }
      dv.x = dws[0]; dv.y = dws[1]; dv.z = dws[2]; dv.w = dws[3];
      int off = mt * 1024 + ln * 32 + half * 16;
      *(uint4*)(sm + L_AFRG + off) = dv;
      *(uint4*)(ws + OFF_ABUF + (size_t)((s + 1) & 1) * 65536 + g * 4096 + off) = dv;
    }
    __syncthreads();

    // ---- projection partial: y[b][s][:] += a_slice @ Wp_slice (waves 0..3) ----
    if (w < 4) {
      v16bf av = *(const v16bf*)(sm + L_AFRG + w * 1024 + lane * 32);
      int mb = (lane >> 4) << 3;
#pragma unroll
      for (int nt = 0; nt < 2; ++nt) {
        v16bf wp = *(const v16bf*)(sm + L_WP + nt * 1024 + lane * 32);
        v8f d = {0.f};
        d = wmma_bf16(av, wp, d);
        int o = nt * 16 + (lane & 15);
#pragma unroll
        for (int r = 0; r < 8; ++r) {
          int b = w * 16 + mb + r;
          atomicAdd(&y[((size_t)b * S_ + s) * OUT_ + o], d[r]);
        }
      }
    }

    // ---- inter-workgroup step barrier ----
    __threadfence();   // make a_next frags + y partials visible at device scope
    __syncthreads();
    if (t == 0) {
      __hip_atomic_fetch_add(&arrive[s], 1u, __ATOMIC_RELEASE, __HIP_MEMORY_SCOPE_AGENT);
      while (__hip_atomic_load(&arrive[s], __ATOMIC_ACQUIRE, __HIP_MEMORY_SCOPE_AGENT) < NWG)
        __builtin_amdgcn_s_sleep(1);
    }
    __builtin_amdgcn_s_cluster_barrier();  // NOP unless cluster-launched (ISA §6.6)
    __syncthreads();
    __threadfence();   // acquire other WGs' a_next before next step's TDM read
  }
}

// ---------------------------------------------------------------------------
extern "C" void kernel_launch(void* const* d_in, const int* in_sizes, int n_in,
                              void* d_out, int out_size, void* d_ws, size_t ws_size,
                              hipStream_t stream) {
  const int*   x   = (const int*)d_in[0];
  const float* emb = (const float*)d_in[1];
  const float* Wf  = (const float*)d_in[2];
  const float* bf  = (const float*)d_in[3];
  const float* Wi  = (const float*)d_in[4];
  const float* bi  = (const float*)d_in[5];
  const float* Wo  = (const float*)d_in[6];
  const float* bo  = (const float*)d_in[7];
  const float* Wc  = (const float*)d_in[8];
  const float* bc  = (const float*)d_in[9];
  const float* Wp  = (const float*)d_in[10];
  const float* bp  = (const float*)d_in[11];
  float* y = (float*)d_out;
  char*  ws = (char*)d_ws;

  (void)in_sizes; (void)n_in; (void)out_size; (void)ws_size;
  (void)hipFuncSetAttribute((const void*)lstm_seq,
                            hipFuncAttributeMaxDynamicSharedMemorySize, LDS_TOTAL);

  lstm_init_y<<<(B_ * S_ * OUT_ + 255) / 256, 256, 0, stream>>>(y, bp);
  lstm_zero_ws<<<(256 + 16384 + 255) / 256, 256, 0, stream>>>(ws);
  lstm_pack_ba<<<(int)(SZ_BA / 4 + 255) / 256, 256, 0, stream>>>(ws, Wf, Wi, Wo, Wc);
  lstm_pack_bx<<<(int)(SZ_BX / 4 + 255) / 256, 256, 0, stream>>>(ws, Wf, Wi, Wo, Wc);
  lstm_pack_wp<<<(int)(SZ_WP / 4 + 255) / 256, 256, 0, stream>>>(ws, Wp);
  lstm_pack_ax<<<(int)(SZ_AX / 4 + 255) / 256, 256, 0, stream>>>(ws, x, emb);
  lstm_seq<<<NWG, 256, LDS_TOTAL, stream>>>(bf, bi, bo, bc, y, ws);
}
